// ProcessorBlock_16655883174348
// MI455X (gfx1250) — matrix operand order repaired and checked by reading, not compiled, not run on hardware
//
#include <hip/hip_runtime.h>

#define D 64
#define DH 128
#define GEN_EPS 1e-7f
#define LN_EPS 1e-5f

typedef __attribute__((ext_vector_type(16))) _Float16 v16h;
typedef __attribute__((ext_vector_type(2)))  _Float16 v2h;
typedef __attribute__((ext_vector_type(8)))  float    v8f;

// ---------------------------------------------------------------- zero init
__global__ void k_zero(float* __restrict__ p, size_t cnt) {
  size_t i = (size_t)blockIdx.x * blockDim.x + threadIdx.x;
  if (i < cnt) p[i] = 0.0f;
}

// ------------------------------------------- one-shot weight f32->f16 convert
// W1h: [128x64] halves, W2h: [64x128] halves, row-major, packed after W1h.
__global__ void k_cvt_w(const float* __restrict__ W1, const float* __restrict__ W2,
                        _Float16* __restrict__ Wh) {
  int i = blockIdx.x * blockDim.x + threadIdx.x;
  if (i < DH * D) {
    Wh[i] = (_Float16)W1[i];
  } else if (i < 2 * DH * D) {
    Wh[i] = (_Float16)W2[i - DH * D];
  }
}

// ------------------------------------------------------- pass 1: segment max
// 16 threads per edge, float4 per thread. All msgs >= eps > 0, so uint-ordered
// atomicMax on the bit pattern is exact, and 0-init == reference's empty->0.
__global__ void k_edge_max(const float* __restrict__ x, const int* __restrict__ ei,
                           const float* __restrict__ ea, unsigned int* __restrict__ maxbuf,
                           int E) {
  long long t = (long long)blockIdx.x * blockDim.x + threadIdx.x;
  long long e = t >> 4;
  if (e >= E) return;
  int c = (int)(t & 15) << 2;
  int src = ei[e];
  int dst = ei[(long long)E + e];
  float4 xv = *(const float4*)(x + (long long)src * D + c);
  float4 av = *(const float4*)(ea + e * (long long)D + c);
  float m0 = fmaxf(xv.x + av.x, 0.0f) + GEN_EPS;
  float m1 = fmaxf(xv.y + av.y, 0.0f) + GEN_EPS;
  float m2 = fmaxf(xv.z + av.z, 0.0f) + GEN_EPS;
  float m3 = fmaxf(xv.w + av.w, 0.0f) + GEN_EPS;
  unsigned int* p = maxbuf + (long long)dst * D + c;
  atomicMax(p + 0, __float_as_uint(m0));
  atomicMax(p + 1, __float_as_uint(m1));
  atomicMax(p + 2, __float_as_uint(m2));
  atomicMax(p + 3, __float_as_uint(m3));
}

// ----------------------------------- pass 2: num = sum(e*msg), den = sum(e)
__global__ void k_edge_sum(const float* __restrict__ x, const int* __restrict__ ei,
                           const float* __restrict__ ea, const float* __restrict__ maxbuf,
                           float* __restrict__ numbuf, float* __restrict__ denbuf, int E) {
  long long t = (long long)blockIdx.x * blockDim.x + threadIdx.x;
  long long e = t >> 4;
  if (e >= E) return;
  int c = (int)(t & 15) << 2;
  int src = ei[e];
  int dst = ei[(long long)E + e];
  float4 xv = *(const float4*)(x + (long long)src * D + c);
  float4 av = *(const float4*)(ea + e * (long long)D + c);
  float4 mx = *(const float4*)(maxbuf + (long long)dst * D + c);
  float m0 = fmaxf(xv.x + av.x, 0.0f) + GEN_EPS;
  float m1 = fmaxf(xv.y + av.y, 0.0f) + GEN_EPS;
  float m2 = fmaxf(xv.z + av.z, 0.0f) + GEN_EPS;
  float m3 = fmaxf(xv.w + av.w, 0.0f) + GEN_EPS;
  float e0 = __expf(m0 - mx.x);
  float e1 = __expf(m1 - mx.y);
  float e2 = __expf(m2 - mx.z);
  float e3 = __expf(m3 - mx.w);
  float* dn = denbuf + (long long)dst * D + c;
  float* nm = numbuf + (long long)dst * D + c;
  unsafeAtomicAdd(dn + 0, e0);
  unsafeAtomicAdd(dn + 1, e1);
  unsafeAtomicAdd(dn + 2, e2);
  unsafeAtomicAdd(dn + 3, e3);
  unsafeAtomicAdd(nm + 0, e0 * m0);
  unsafeAtomicAdd(nm + 1, e1 * m1);
  unsafeAtomicAdd(nm + 2, e2 * m2);
  unsafeAtomicAdd(nm + 3, e3 * m3);
}

// -------------------------------------------------------------- WMMA helpers
__device__ inline v8f wmma_f16(v16h a, v16h b, v8f c) {
  return __builtin_amdgcn_wmma_f32_16x16x32_f16(false, a, false, b, (short)0, c,
                                                false, false);
}

// A-fragment (16-bit A 16x32): lane L: M = L%16; halves = 16 consecutive K
// (K base +16 for the upper lane half), matching the B-fragment order.
__device__ inline v16h load_a_frag(const _Float16* __restrict__ base, int ldm,
                                   int kt, int lane) {
  v16h a;
  int m = lane & 15;
  int koff = kt * 32 + ((lane & 16) ? 16 : 0);
#pragma unroll
  for (int v = 0; v < 8; ++v) {
    int k = koff + 2 * v;
    a[2 * v]     = base[m * ldm + k];
    a[2 * v + 1] = base[m * ldm + k + 1];
  }
  return a;
}

// B-fragment (16-bit B KxN=32x16): lane L: N = L%16, halves = 16 consecutive K
// (K base +16 for hi lane half). Wh is pre-converted f16 row-major [Nout, K];
// each lane reads 32 contiguous, 32B-aligned bytes -> two global_load_b128.
__device__ inline v16h load_b_frag(const _Float16* __restrict__ Wh, int K,
                                   int ntile, int kt, int lane) {
  const _Float16* p =
      Wh + (ntile * 16 + (lane & 15)) * K + kt * 32 + ((lane & 16) ? 16 : 0);
  return *(const v16h*)p;
}

// --------------------------- fused: agg = num/den, out = agg + x, MLP via WMMA
__global__ __launch_bounds__(256) void k_mlp(
    const float* __restrict__ x, const float* __restrict__ numbuf,
    const float* __restrict__ denbuf, const _Float16* __restrict__ W1h,
    const float* __restrict__ b1, const float* __restrict__ g1,
    const float* __restrict__ be1, const _Float16* __restrict__ W2h,
    const float* __restrict__ b2, float* __restrict__ out, int n) {
  __shared__ _Float16 ldsA[8 * 16 * D];   // 16 KB: per-wave input tile  (16x64)
  __shared__ _Float16 ldsH[8 * 16 * DH];  // 32 KB: per-wave hidden tile (16x128)

  const int lane = threadIdx.x & 31;
  const int wave = threadIdx.x >> 5;
  const int rowBase = (blockIdx.x * 8 + wave) * 16;
  _Float16* A = ldsA + wave * 16 * D;
  _Float16* H = ldsH + wave * 16 * DH;

  // Stage a = num/den + x (f16) into LDS, pair-packed b32 stores.
  for (int i = lane; i < 16 * D / 2; i += 32) {
    int r = i >> 5;               // 32 half-pairs per 64-wide row
    int c = (i & 31) << 1;
    int row = rowBase + r;
    row = (row < n) ? row : (n - 1);
    long long g = (long long)row * D + c;
    float d0 = denbuf[g], d1 = denbuf[g + 1];
    float a0 = ((d0 > 0.0f) ? (numbuf[g] / d0) : 0.0f) + x[g];
    float a1 = ((d1 > 0.0f) ? (numbuf[g + 1] / d1) : 0.0f) + x[g + 1];
    v2h pk;
    pk[0] = (_Float16)a0;
    pk[1] = (_Float16)a1;
    ((v2h*)A)[i] = pk;
  }
  __syncthreads();

  // ---- GEMM1: [16x64] x [64x128] -> acc[8] (N-tiles), K in 2 steps of 32
  v16h a0 = load_a_frag(A, D, 0, lane);
  v16h a1 = load_a_frag(A, D, 1, lane);
  v8f acc[8];
#pragma unroll
  for (int nt = 0; nt < 8; ++nt) {
    v8f c = {};
    c = wmma_f16(a0, load_b_frag(W1h, D, nt, 0, lane), c);
    c = wmma_f16(a1, load_b_frag(W1h, D, nt, 1, lane), c);
    float bb = b1[nt * 16 + (lane & 15)];
#pragma unroll
    for (int r = 0; r < 8; ++r) c[r] += bb;
    acc[nt] = c;
  }

  // Per-lane gamma/beta for the 8 N-tiles (N = nt*16 + lane%16)
  float g1v[8], be1v[8];
#pragma unroll
  for (int nt = 0; nt < 8; ++nt) {
    int nn = nt * 16 + (lane & 15);
    g1v[nt] = g1[nn];
    be1v[nt] = be1[nn];
  }

  // ---- LayerNorm(128) + ReLU in C-layout; row m = r + 8*(lane>=16), N = lane%16
#pragma unroll
  for (int r = 0; r < 8; ++r) {
    float sv = 0.0f, sq = 0.0f;
#pragma unroll
    for (int nt = 0; nt < 8; ++nt) {
      float h = acc[nt][r];
      sv += h;
      sq += h * h;
    }
#pragma unroll
    for (int m_ = 8; m_ >= 1; m_ >>= 1) {
      sv += __shfl_xor(sv, m_, 16);
      sq += __shfl_xor(sq, m_, 16);
    }
    float mu = sv * (1.0f / 128.0f);
    float var = sq * (1.0f / 128.0f) - mu * mu;
    float rs = rsqrtf(var + LN_EPS);
    int mrow = r + ((lane & 16) ? 8 : 0);
#pragma unroll
    for (int nt = 0; nt < 8; ++nt) {
      float hn = (acc[nt][r] - mu) * rs * g1v[nt] + be1v[nt];
      hn = fmaxf(hn, 0.0f);
      H[mrow * DH + nt * 16 + (lane & 15)] = (_Float16)hn;
    }
  }
  __syncthreads();

  // ---- GEMM2: [16x128] x [128x64] -> 4 N-tiles, K in 4 steps of 32
  v16h hf[4];
#pragma unroll
  for (int kt = 0; kt < 4; ++kt) hf[kt] = load_a_frag(H, DH, kt, lane);

#pragma unroll
  for (int nt = 0; nt < 4; ++nt) {
    v8f c = {};
#pragma unroll
    for (int kt = 0; kt < 4; ++kt)
      c = wmma_f16(hf[kt], load_b_frag(W2h, DH, nt, kt, lane), c);
    int ncol = nt * 16 + (lane & 15);
    float bb = b2[ncol];
    int mbase = rowBase + ((lane & 16) ? 8 : 0);
#pragma unroll
    for (int r = 0; r < 8; ++r) {
      int row = mbase + r;
      if (row < n) out[(long long)row * D + ncol] = c[r] + bb;
    }
  }
}

// ---------------------------------------------------------------------------
extern "C" void kernel_launch(void* const* d_in, const int* in_sizes, int n_in,
                              void* d_out, int out_size, void* d_ws, size_t ws_size,
                              hipStream_t stream) {
  const float* x   = (const float*)d_in[0];
  const int*   ei  = (const int*)d_in[1];
  const float* ea  = (const float*)d_in[2];
  const float* W1  = (const float*)d_in[3];
  const float* b1  = (const float*)d_in[4];
  const float* g1  = (const float*)d_in[5];
  const float* be1 = (const float*)d_in[6];
  const float* W2  = (const float*)d_in[7];
  const float* b2  = (const float*)d_in[8];
  float* out = (float*)d_out;

  int n = in_sizes[0] / D;   // 50000
  int E = in_sizes[1] / 2;   // 800000

  float* maxbuf = (float*)d_ws;
  float* numbuf = maxbuf + (size_t)n * D;
  float* denbuf = numbuf + (size_t)n * D;
  _Float16* Wh  = (_Float16*)(denbuf + (size_t)n * D);  // W1h then W2h (32 KB)

  // Zero reduction buffers + convert weights (independent; stream-ordered).
  size_t initCnt = (size_t)n * D * 3;
  k_zero<<<(int)((initCnt + 511) / 512), 512, 0, stream>>>(maxbuf, initCnt);
  k_cvt_w<<<(2 * DH * D + 255) / 256, 256, 0, stream>>>(W1, W2, Wh);

  long long tE = (long long)E * 16;
  int gE = (int)((tE + 255) / 256);
  k_edge_max<<<gE, 256, 0, stream>>>(x, ei, ea, (unsigned int*)maxbuf, E);
  k_edge_sum<<<gE, 256, 0, stream>>>(x, ei, ea, maxbuf, numbuf, denbuf, E);

  k_mlp<<<(n + 127) / 128, 256, 0, stream>>>(x, numbuf, denbuf, Wh, b1, g1, be1,
                                             Wh + DH * D, b2, out, n);
}
